// TGAT_36352603193784
// MI455X (gfx1250) — matrix operand order repaired and checked by reading, not compiled, not run on hardware
//
#include <hip/hip_runtime.h>

// ---------------------------------------------------------------------------
// TGAT fused attention for MI455X (gfx1250, wave32, WMMA f16->f32).
// Key algebraic optimization: since L=1, fold Wk/Wv through the attention:
//   attn = (q_h @ Wk_h^T) . kv_row      (project q into KD space, 20x fewer FLOPs)
//   out  = (scores @ kv_rows) @ Wv_h    (weighted-sum raw kv first)
// This makes the kernel HBM-bound (~620MB @ 23.3TB/s ~ 27us); WMMA keeps the
// remaining ~14 GFLOP off the VALU critical path.
// ---------------------------------------------------------------------------

namespace {
constexpr int kNF = 172, kTF = 100;
constexpr int kQD = 272, kKD = 444, kN = 20;
constexpr int NPB = 8;              // nodes per workgroup
constexpr int KVR = NPB * kN;       // 160 kv rows per block
constexpr int KDP = 448;            // KD padded to 14*32
constexpr int KVS = 456;            // kv LDS row stride (halves), bank-skewed
constexpr int QDP = 288;            // QD padded to 9*32
constexpr int DHP = 160;            // DH padded to 5*32
// f16 weight workspace offsets (in halves)
constexpr int oWq = 0;                        // WqT  [272][288]
constexpr int oWk = oWq + kQD * QDP;          // WkP  [448][2*160]
constexpr int oWv = oWk + KDP * (2 * DHP);    // WvT  [2][144][448]
constexpr int oWr = oWv + 2 * 144 * KDP;      // WrT  [272][288]
constexpr int kWsHalves = oWr + kQD * QDP;    // 429056 halves = 858112 B
}

typedef _Float16 h16v __attribute__((ext_vector_type(16)));
typedef _Float16 h8v  __attribute__((ext_vector_type(8)));
typedef _Float16 h4v  __attribute__((ext_vector_type(4)));
typedef _Float16 h2v  __attribute__((ext_vector_type(2)));
typedef float    f8v  __attribute__((ext_vector_type(8)));

// A-matrix fragment, 16x32 f16 (ISA 7.12.2): lane m=lane&15, half=lane>>4,
// elems 0..7 -> K = half*8 + e ; elems 8..15 -> K = 16 + half*8 + (e-8).
static __device__ __forceinline__ h16v frag_A(const _Float16* base, int stride, int lane) {
  const int m = lane & 15, hf = lane >> 4;
  const _Float16* p = base + m * stride + hf * 8;
  h8v lo = *(const h8v*)(p);
  h8v hi = *(const h8v*)(p + 16);
  return __builtin_shufflevector(lo, hi, 0, 1, 2, 3, 4, 5, 6, 7, 8, 9, 10, 11, 12, 13, 14, 15);
}
// B-matrix fragment, 32x16 f16: lane n=lane&15 (N column), half=lane>>4,
// elems 0..7 -> K = half*8 + e ; elems 8..15 -> K = 16 + half*8 + (e-8).  B stored as [N][K] row-major with given stride.
static __device__ __forceinline__ h16v frag_B(const _Float16* base, int stride, int lane) {
  const int n = lane & 15, hf = lane >> 4;
  const _Float16* p = base + n * stride + hf * 8;
  h8v lo = *(const h8v*)(p);
  h8v hi = *(const h8v*)(p + 16);
  return __builtin_shufflevector(lo, hi, 0, 1, 2, 3, 4, 5, 6, 7, 8, 9, 10, 11, 12, 13, 14, 15);
}

#define WMMA_F16(acc, a, b) \
  __builtin_amdgcn_wmma_f32_16x16x32_f16(false, (a), false, (b), (short)0, (acc), false, false)

// ---------------------------------------------------------------------------
// Prep: convert fp32 weights to f16 in WMMA-friendly (transposed/padded) form.
//   WqT[n][k]        = Wq[k][n]            (k<272)        [272][288]
//   WkP[kd][h*160+d] = Wk[kd][h*136+d]     (kd<444,d<136) [448][320]
//   WvT[h][d][kd]    = Wv[kd][h*136+d]     (kd<444,d<136) [2][144][448]
//   WrT[n][k]        = Wr[k][n]            (k<272)        [272][288]
// ---------------------------------------------------------------------------
__global__ __launch_bounds__(256) void tgat_prep_weights(
    const float* __restrict__ Wq, const float* __restrict__ Wk,
    const float* __restrict__ Wv, const float* __restrict__ Wr,
    _Float16* __restrict__ ws) {
  int i = blockIdx.x * 256 + threadIdx.x;
  if (i < kQD * QDP) {
    int n = i / QDP, k = i % QDP;
    ws[oWq + i] = (_Float16)((k < kQD) ? Wq[k * kQD + n] : 0.0f);
    return;
  }
  i -= kQD * QDP;
  if (i < KDP * 2 * DHP) {
    int kd = i / (2 * DHP), c = i % (2 * DHP);
    int hh = c / DHP, d = c % DHP;
    float v = (kd < kKD && d < 136) ? Wk[kd * kQD + hh * 136 + d] : 0.0f;
    ws[oWk + i] = (_Float16)v;
    return;
  }
  i -= KDP * 2 * DHP;
  if (i < 2 * 144 * KDP) {
    int hh = i / (144 * KDP), r = i % (144 * KDP);
    int d = r / KDP, kd = r % KDP;
    float v = (kd < kKD && d < 136) ? Wv[kd * kQD + hh * 136 + d] : 0.0f;
    ws[oWv + i] = (_Float16)v;
    return;
  }
  i -= 2 * 144 * KDP;
  if (i < kQD * QDP) {
    int n = i / QDP, k = i % QDP;
    ws[oWr + i] = (_Float16)((k < kQD) ? Wr[k * kQD + n] : 0.0f);
  }
}

// ---------------------------------------------------------------------------
// Fused per-8-node kernel. 256 threads = 8 wave32.
// ---------------------------------------------------------------------------
__global__ __launch_bounds__(256) void tgat_fused(
    const float* __restrict__ nodef, const float* __restrict__ nodet,
    const float* __restrict__ nnf, const float* __restrict__ ntf,
    const float* __restrict__ nef, const int* __restrict__ nmask,
    const _Float16* __restrict__ wbuf, const float* __restrict__ br,
    const float* __restrict__ gam, const float* __restrict__ bet,
    float* __restrict__ out1, float* __restrict__ out2) {
  __shared__ _Float16 lds_kv[KVR * KVS];        // 145920 B, raw kv features
  __shared__ _Float16 lds_q[16 * QDP];          // q_in (rows 8..15 zero)
  __shared__ _Float16 lds_qh[2 * 16 * DHP];     // per-head projected q
  __shared__ _Float16 lds_t[2 * 16 * KVS];      // t = q_h @ Wk_h^T
  __shared__ _Float16 lds_ctx[2 * 16 * KVS];    // ctx = scores @ kv
  __shared__ _Float16 lds_a2[16 * QDP];         // attention output (concat heads)
  __shared__ float lds_res[NPB * kQD];
  __shared__ float lds_y[NPB * kQD];
  __shared__ float lds_attn[2 * NPB * 24];
  __shared__ int lds_mask[KVR];

  const int tid = threadIdx.x;
  const int lane = tid & 31;
  const int wave = tid >> 5;
  const int blk = blockIdx.x;
  const _Float16* wqT = wbuf + oWq;
  const _Float16* wkP = wbuf + oWk;
  const _Float16* wvT = wbuf + oWv;
  const _Float16* wrT = wbuf + oWr;

  // ---- S0: stage inputs into LDS as f16 -------------------------------------
  // kv tile: concat(nbr_node[172], nbr_edge[172], nbr_time[100]) -> 444, pad 448+
  // boundaries 172/344/444 are all multiples of 4 -> float4 chunks never split.
  for (int i = tid; i < KVR * (KVS / 4); i += 256) {
    const int r = i / (KVS / 4), qc = i % (KVS / 4);
    const int e = blk * KVR + r;  // global (node,neighbor) row
    float4 v;
    if (qc < 43)       v = ((const float4*)(nnf + (size_t)e * kNF))[qc];
    else if (qc < 86)  v = ((const float4*)(nef + (size_t)e * kNF))[qc - 43];
    else if (qc < 111) v = ((const float4*)(ntf + (size_t)e * kTF))[qc - 86];
    else               v = make_float4(0.f, 0.f, 0.f, 0.f);
    h4v hv = {(_Float16)v.x, (_Float16)v.y, (_Float16)v.z, (_Float16)v.w};
    *(h4v*)(lds_kv + r * KVS + qc * 4) = hv;
  }
  // q_in tile + fp32 residual
  for (int i = tid; i < 16 * QDP; i += 256) {
    const int m = i / QDP, c = i % QDP;
    float v = 0.0f;
    if (m < NPB && c < kQD) {
      const int b = blk * NPB + m;
      v = (c < kNF) ? nodef[(size_t)b * kNF + c] : nodet[(size_t)b * kTF + (c - kNF)];
      lds_res[m * kQD + c] = v;
    }
    lds_q[i] = (_Float16)v;
  }
  if (tid < KVR) lds_mask[tid] = nmask[(size_t)blk * KVR + tid];
  for (int i = tid; i < 2 * 16 * DHP; i += 256) lds_qh[i] = (_Float16)0.0f;
  for (int i = tid; i < 2 * 16 * KVS; i += 256) lds_ctx[i] = (_Float16)0.0f;
  for (int i = tid; i < 16 * QDP; i += 256) lds_a2[i] = (_Float16)0.0f;
  __syncthreads();

  // ---- S1: q = q_in @ Wq   [16,288]x[288,272] -------------------------------
  for (int t = wave; t < 17; t += 8) {
    f8v acc = {};
    const _Float16* Bb = wqT + t * 16 * QDP;
#pragma unroll
    for (int kk = 0; kk < 9; ++kk) {
      h16v a = frag_A(lds_q + kk * 32, QDP, lane);
      h16v b = frag_B(Bb + kk * 32, QDP, lane);
      acc = WMMA_F16(acc, a, b);
    }
    const int n = lane & 15, hf = lane >> 4;
    const int c = t * 16 + n, hh = c / 136, d = c - hh * 136;
#pragma unroll
    for (int r = 0; r < 8; ++r) {
      const int m = r + 8 * hf;
      lds_qh[(hh * 16 + m) * DHP + d] = (_Float16)acc[r];
    }
  }
  __syncthreads();

  // ---- S2: t_h = q_h @ Wk_h^T   [16,160]x[160,448] per head -----------------
  for (int t = wave; t < 56; t += 8) {
    const int hh = t / 28, nt = t % 28;
    f8v acc = {};
    const _Float16* Ab = lds_qh + hh * 16 * DHP;
    const _Float16* Bb = wkP + (nt * 16) * (2 * DHP) + hh * DHP;
#pragma unroll
    for (int kk = 0; kk < 5; ++kk) {
      h16v a = frag_A(Ab + kk * 32, DHP, lane);
      h16v b = frag_B(Bb + kk * 32, 2 * DHP, lane);
      acc = WMMA_F16(acc, a, b);
    }
    const int n = lane & 15, hf = lane >> 4;
    const int kd = nt * 16 + n;
#pragma unroll
    for (int r = 0; r < 8; ++r) {
      const int m = r + 8 * hf;
      lds_t[(hh * 16 + m) * KVS + kd] = (_Float16)acc[r];
    }
  }
  __syncthreads();

  // ---- S3: attn = kv @ t_h^T   [160,448]x[448,16]; keep diagonal blocks -----
  for (int t = wave; t < 20; t += 8) {
    const int hh = t / 10, mt = t % 10;
    f8v acc = {};
    const _Float16* Ab = lds_kv + (mt * 16) * KVS;
    const _Float16* Bb = lds_t + hh * 16 * KVS;
#pragma unroll
    for (int kk = 0; kk < 14; ++kk) {
      h16v a = frag_A(Ab + kk * 32, KVS, lane);
      h16v b = frag_B(Bb + kk * 32, KVS, lane);
      acc = WMMA_F16(acc, a, b);
    }
    const int n = lane & 15, hf = lane >> 4;
#pragma unroll
    for (int r = 0; r < 8; ++r) {
      const int g = mt * 16 + r + 8 * hf;   // kv row in [0,160)
      const int node = g / kN, nbr = g - node * kN;
      if (n == node) {
        float v = acc[r] * 0.08574929257125442f;  // 1/sqrt(136)
        if (lds_mask[g] == 0) v = -1e10f;
        lds_attn[(hh * NPB + node) * 24 + nbr] = v;
      }
    }
  }
  __syncthreads();

  // ---- S4a: softmax (16 rows of 20) + scores output -------------------------
  if (tid < 2 * NPB) {
    const int hh = tid >> 3, node = tid & 7;
    float* row = lds_attn + (hh * NPB + node) * 24;
    float mx = row[0];
    for (int n = 1; n < kN; ++n) mx = fmaxf(mx, row[n]);
    float s = 0.0f;
    for (int n = 0; n < kN; ++n) { float e = __expf(row[n] - mx); row[n] = e; s += e; }
    const float inv = 1.0f / s;
    const int bg = blk * NPB + node;
    for (int n = 0; n < kN; ++n) {
      const float p = row[n] * inv;
      row[n] = p;
      out2[(size_t)bg * (2 * kN) + hh * kN + n] = p;
    }
  }
  __syncthreads();

  // ---- S4b: ctx_h[node] = sum_n s[n] * kv[node*20+n]  (VALU, row-contig) ----
  for (int i = tid; i < 2 * NPB * (KDP / 2); i += 256) {
    const int hh = i / (NPB * (KDP / 2));
    const int rem = i - hh * (NPB * (KDP / 2));
    const int node = rem / (KDP / 2);
    const int kd = (rem - node * (KDP / 2)) * 2;
    const float* srow = lds_attn + (hh * NPB + node) * 24;
    float a0 = 0.0f, a1 = 0.0f;
#pragma unroll 4
    for (int n = 0; n < kN; ++n) {
      h2v kvp = *(const h2v*)(lds_kv + (node * kN + n) * KVS + kd);
      const float s = srow[n];
      a0 += s * (float)kvp[0];
      a1 += s * (float)kvp[1];
    }
    h2v o = {(_Float16)a0, (_Float16)a1};
    *(h2v*)(lds_ctx + (hh * 16 + node) * KVS + kd) = o;
  }
  __syncthreads();

  // ---- S5: out_h = ctx_h @ Wv_h   [16,448]x[448,136] ------------------------
  for (int t = wave; t < 18; t += 8) {
    const int hh = t / 9, nt = t % 9;
    f8v acc = {};
    const _Float16* Ab = lds_ctx + hh * 16 * KVS;
    const _Float16* Bb = wvT + (hh * 144 + nt * 16) * KDP;
#pragma unroll
    for (int kk = 0; kk < 14; ++kk) {
      h16v a = frag_A(Ab + kk * 32, KVS, lane);
      h16v b = frag_B(Bb + kk * 32, KDP, lane);
      acc = WMMA_F16(acc, a, b);
    }
    const int n = lane & 15, hf = lane >> 4;
    const int c = nt * 16 + n;
    if (c < 136) {
#pragma unroll
      for (int r = 0; r < 8; ++r) {
        const int m = r + 8 * hf;
        lds_a2[m * QDP + hh * 136 + c] = (_Float16)acc[r];
      }
    }
  }
  __syncthreads();

  // ---- S6: y = a2 @ Wr + br + residual --------------------------------------
  for (int t = wave; t < 17; t += 8) {
    f8v acc = {};
    const _Float16* Bb = wrT + t * 16 * QDP;
#pragma unroll
    for (int kk = 0; kk < 9; ++kk) {
      h16v a = frag_A(lds_a2 + kk * 32, QDP, lane);
      h16v b = frag_B(Bb + kk * 32, QDP, lane);
      acc = WMMA_F16(acc, a, b);
    }
    const int n = lane & 15, hf = lane >> 4;
    const int c = t * 16 + n;
    const float bias = br[c];
#pragma unroll
    for (int r = 0; r < 8; ++r) {
      const int m = r + 8 * hf;
      if (m < NPB) lds_y[m * kQD + c] = acc[r] + bias + lds_res[m * kQD + c];
    }
  }
  __syncthreads();

  // ---- S7: LayerNorm, wave w handles node row w -----------------------------
  {
    const int row = wave;  // 0..7
    float s = 0.0f, s2 = 0.0f;
    for (int c = lane; c < kQD; c += 32) {
      const float v = lds_y[row * kQD + c];
      s += v;
      s2 += v * v;
    }
#pragma unroll
    for (int off = 16; off > 0; off >>= 1) {
      s += __shfl_xor(s, off, 32);
      s2 += __shfl_xor(s2, off, 32);
    }
    const float mu = s * (1.0f / kQD);
    const float var = s2 * (1.0f / kQD) - mu * mu;
    const float inv = rsqrtf(var + 1e-5f);
    const int bg = blk * NPB + row;
    for (int c = lane; c < kQD; c += 32) {
      const float v = (lds_y[row * kQD + c] - mu) * inv;
      out1[(size_t)bg * kQD + c] = gam[c] * v + bet[c];
    }
  }
}

extern "C" void kernel_launch(void* const* d_in, const int* in_sizes, int n_in,
                              void* d_out, int out_size, void* d_ws, size_t ws_size,
                              hipStream_t stream) {
  const float* nodef = (const float*)d_in[0];
  const float* nodet = (const float*)d_in[1];
  const float* nnf = (const float*)d_in[2];
  const float* ntf = (const float*)d_in[3];   // neighbor time (before edge in dict!)
  const float* nef = (const float*)d_in[4];   // neighbor edge
  const int* nmask = (const int*)d_in[5];
  const float* Wq = (const float*)d_in[6];
  const float* Wk = (const float*)d_in[7];
  const float* Wv = (const float*)d_in[8];
  const float* Wr = (const float*)d_in[9];
  const float* br = (const float*)d_in[10];
  const float* gam = (const float*)d_in[11];
  const float* bet = (const float*)d_in[12];

  const int Bn = in_sizes[0] / kNF;           // 16384
  _Float16* wbuf = (_Float16*)d_ws;
  float* out1 = (float*)d_out;
  float* out2 = out1 + (size_t)Bn * kQD;

  tgat_prep_weights<<<(kWsHalves + 255) / 256, 256, 0, stream>>>(Wq, Wk, Wv, Wr, wbuf);
  tgat_fused<<<Bn / NPB, 256, 0, stream>>>(nodef, nodet, nnf, ntf, nef, nmask, wbuf,
                                           br, gam, bet, out1, out2);
}
